// ConvLayer_5342939317068
// MI455X (gfx1250) — matrix operand order repaired and checked by reading, not compiled, not run on hardware
//
#include <hip/hip_runtime.h>
#include <math.h>

// ---------------------------------------------------------------------------
// CDNA5 (gfx1250) fused CGCNN attention conv layer.
// All GEMMs use v_wmma_f32_16x16x32_bf16 (wave32, 16x16 tiles, f32 accum).
// Loop unrolling is deliberately bounded so the per-wave live set stays under
// the VGPR budget (round-2 feedback showed scratch spill/reload of weight
// fragments when the whole pipeline was one straight-line region).
// ---------------------------------------------------------------------------

typedef __bf16 bf16_t;
typedef __attribute__((ext_vector_type(16))) __bf16 v16bf;
typedef __attribute__((ext_vector_type(8)))  float  v8f;

#define DEVINL __device__ __forceinline__

union AB16 { uint4 q[2]; v16bf v; };

DEVINL v8f vzero8() {
  v8f z = {0.f, 0.f, 0.f, 0.f, 0.f, 0.f, 0.f, 0.f};
  return z;
}

// A-matrix fragment (16x32 bf16, rows contiguous in p with element stride
// `stride`, stride must be a multiple of 8 elements for 16B alignment).
// Layout: lanes0-15 row=lane, v0..7=K(k0..k0+15)
// with k0 = kbase (lanes<16) or kbase+16 (lanes>=16).
DEVINL v16bf frag_a16(const bf16_t* p, int stride, int kbase, int lane) {
  int row = lane & 15;
  int k0  = kbase + ((lane & 16) ? 16 : 0);
  const bf16_t* r0 = p + row * stride + k0;
  AB16 f;
  f.q[0] = *((const uint4*)(r0));
  f.q[1] = *((const uint4*)(r0 + 8));
  return f.v;
}

// B-matrix fragment (32x16 bf16) where B(k,n) = p[(nbase+n)*stride + kbase+k]
// i.e. p holds B^T with rows contiguous (weight matrices stored [out][in]).
// Layout: lane holds column n=lane&15; lanes0-15 K=0..15, lanes16-31 K=16..31.
DEVINL v16bf frag_b_rows(const bf16_t* p, int stride, int nbase, int kbase, int lane) {
  int n0 = lane & 15;
  int k0 = kbase + ((lane & 16) ? 16 : 0);
  const bf16_t* r0 = p + (nbase + n0) * stride + k0;
  AB16 f;
  f.q[0] = *((const uint4*)(r0));
  f.q[1] = *((const uint4*)(r0 + 8));
  return f.v;
}

// B-matrix fragment gathered column-wise: B(k,n) = p[k*stride + nbase+n],
// p has only 16 valid rows so K=16..31 (upper lanes) are zero.
DEVINL v16bf frag_b_cols16(const bf16_t* p, int stride, int nbase, int lane) {
  v16bf v;
  if (lane < 16) {
    int n0 = nbase + lane;
#pragma unroll
    for (int i = 0; i < 16; ++i) v[i] = p[i * stride + n0];
  } else {
#pragma unroll
    for (int i = 0; i < 16; ++i) v[i] = (bf16_t)0.0f;
  }
  return v;
}

// D(16x16 f32) -> bf16 store. D layout: VGPR r = row (r | r+8), col = lane&15.
DEVINL void store_d(bf16_t* p, int stride, int nbase, v8f acc, int lane) {
  int col = nbase + (lane & 15);
  int rb  = (lane & 16) ? 8 : 0;
#pragma unroll
  for (int r = 0; r < 8; ++r) p[(rb + r) * stride + col] = (bf16_t)acc[r];
}

DEVINL v8f wmma_bf16(v16bf a, v16bf b, v8f c) {
  return __builtin_amdgcn_wmma_f32_16x16x32_bf16(false, a, false, b, (short)0, c,
                                                 false, false);
}

// ---------------------------------------------------------------------------
// Phase 0: compute gated pre-activations, accumulate BN1 channel sums.
// Phase 1: recompute, apply BN1, gate (sigmoid*softplus), sum over the 12
//          neighbor slots, write nbr_sumed, accumulate BN2 channel sums.
// One wave = one atom per iteration. Requires N % (2*gridDim.x) == 0.
// ---------------------------------------------------------------------------
__global__ __launch_bounds__(64) void conv_attn_fused(
    const float* __restrict__ atom_in, const float* __restrict__ nbr_fea,
    const int* __restrict__ nbr_idx,
    const bf16_t* __restrict__ WKb, const bf16_t* __restrict__ WQb,
    const bf16_t* __restrict__ WVb, const bf16_t* __restrict__ WOb,
    const bf16_t* __restrict__ Wfcb, const float* __restrict__ b_fc,
    const float* __restrict__ bn1_scale, const float* __restrict__ bn1_shift,
    float* __restrict__ stats_out, float* __restrict__ nbr_out,
    int N, int phase)
{
  constexpr int TS = 264;  // stride of 16x256 h-buffer / 16x192 total-buffer
  constexpr int KS = 72;   // stride of 16x64 K/Q/V buffers
  __shared__ bf16_t sTH[2][16 * 264];   // total (192 cols) then h (256 cols)
  __shared__ bf16_t sKm[2][16 * 72];
  __shared__ bf16_t sQm[2][16 * 72];    // Q, then reused for attn output
  __shared__ bf16_t sVm[2][16 * 72];
  __shared__ bf16_t sWm[2][16 * 32];    // softmax weights, K padded to 32
  __shared__ float  sStat[256];

  const int tid  = threadIdx.x;
  const int wave = tid >> 5;
  const int lane = tid & 31;
  const int col  = lane & 15;

  for (int i = tid; i < 256; i += 64) sStat[i] = 0.f;
  for (int i = lane; i < 16 * 32; i += 32) sWm[wave][i] = (bf16_t)0.0f;
  __syncthreads();

  bf16_t* tot = sTH[wave];
  bf16_t* sK  = sKm[wave];
  bf16_t* sQ  = sQm[wave];
  bf16_t* sV  = sVm[wave];
  bf16_t* sW  = sWm[wave];

  // K/Q/V projection, specialized per weight pointer (concrete address
  // spaces). Column-tile loop kept rolled to bound VGPR pressure; the 6-step
  // k-chain is unrolled for back-to-back WMMAs.
  auto proj_192_64 = [&](const bf16_t* __restrict__ W, bf16_t* out) {
#pragma unroll 1
    for (int t = 0; t < 4; ++t) {
      v8f acc = vzero8();
#pragma unroll
      for (int kk = 0; kk < 6; ++kk)
        acc = wmma_bf16(frag_a16(tot, TS, 32 * kk, lane),
                        frag_b_rows(W, 192, 16 * t, 32 * kk, lane), acc);
      store_d(out, KS, 16 * t, acc, lane);
    }
  };

  for (int n = blockIdx.x * 2 + wave; n < N; n += gridDim.x * 2) {
    // ---- build total = [self | gathered neighbor | edge] (12x192), pad to 16
#pragma unroll 1
    for (int i = lane; i < 12 * 192; i += 32) {
      int m = i / 192, c = i - m * 192;
      float x;
      if (c < 64)       x = atom_in[n * 64 + c];
      else if (c < 128) x = atom_in[nbr_idx[n * 12 + m] * 64 + (c - 64)];
      else              x = nbr_fea[(n * 12 + m) * 64 + (c - 128)];
      tot[m * TS + c] = (bf16_t)x;
    }
#pragma unroll 1
    for (int i = lane; i < 4 * 192; i += 32) {
      int m = 12 + i / 192, c = i % 192;
      tot[m * TS + c] = (bf16_t)0.0f;
    }
    __builtin_amdgcn_wave_barrier();

    // ---- K,Q,V = total @ W^T   (3 x 4 col-tiles x 6 k-steps of WMMA)
    proj_192_64(WKb, sK);
    proj_192_64(WQb, sQ);
    proj_192_64(WVb, sV);
    __builtin_amdgcn_wave_barrier();

    // ---- logits = K @ Q^T / 8, softmax over 12 valid columns
    {
      v8f lg = vzero8();
      lg = wmma_bf16(frag_a16(sK, KS, 0, lane),  frag_b_rows(sQ, KS, 0, 0, lane),  lg);
      lg = wmma_bf16(frag_a16(sK, KS, 32, lane), frag_b_rows(sQ, KS, 0, 32, lane), lg);
#pragma unroll
      for (int r = 0; r < 8; ++r) {
        float v = lg[r] * 0.125f;            // 1/sqrt(64)
        if (col >= 12) v = -1e30f;           // mask padded columns
        float mx = v;
        mx = fmaxf(mx, __shfl_xor(mx, 1));
        mx = fmaxf(mx, __shfl_xor(mx, 2));
        mx = fmaxf(mx, __shfl_xor(mx, 4));
        mx = fmaxf(mx, __shfl_xor(mx, 8));   // row max within 16-lane half
        float ev = __expf(v - mx);
        float sm = ev;
        sm += __shfl_xor(sm, 1);
        sm += __shfl_xor(sm, 2);
        sm += __shfl_xor(sm, 4);
        sm += __shfl_xor(sm, 8);
        float w = ev / sm;
        if ((lane & 16) && r >= 4) w = 0.f;  // zero padded rows 12..15
        sW[((lane & 16) ? 8 + r : r) * 32 + col] = (bf16_t)w;
      }
    }
    __builtin_amdgcn_wave_barrier();

    // ---- attn = w @ V  (K padded 12->32; result overwrites sQ)
    {
      v16bf aw = frag_a16(sW, 32, 0, lane);
#pragma unroll 1
      for (int t = 0; t < 4; ++t) {
        v8f acc = vzero8();
        acc = wmma_bf16(aw, frag_b_cols16(sV, KS, 16 * t, lane), acc);
        store_d(sQ, KS, 16 * t, acc, lane);
      }
    }
    __builtin_amdgcn_wave_barrier();

    // ---- h = attn @ WO^T  (16 col-tiles x 2 k-steps; overwrites tot)
    {
      v16bf a0 = frag_a16(sQ, KS, 0, lane);
      v16bf a1 = frag_a16(sQ, KS, 32, lane);
#pragma unroll 1
      for (int t = 0; t < 16; ++t) {
        v8f acc = vzero8();
        acc = wmma_bf16(a0, frag_b_rows(WOb, 64, 16 * t, 0, lane), acc);
        acc = wmma_bf16(a1, frag_b_rows(WOb, 64, 16 * t, 32, lane), acc);
        store_d(tot, TS, 16 * t, acc, lane);
      }
    }
    __builtin_amdgcn_wave_barrier();

    // ---- gated = h @ Wfc^T  (8 col-tiles x 8 k-steps, accums in registers).
    // k-loop rolled; tile loop split in two unrolled halves so at most 4
    // weight fragments (32 VGPRs) are in flight above the 64-VGPR accumulators.
    v8f g[8];
#pragma unroll
    for (int t = 0; t < 8; ++t) g[t] = vzero8();
#pragma unroll 1
    for (int kk = 0; kk < 8; ++kk) {
      v16bf a = frag_a16(tot, TS, 32 * kk, lane);
#pragma unroll
      for (int t = 0; t < 4; ++t)
        g[t] = wmma_bf16(a, frag_b_rows(Wfcb, 256, 16 * t, 32 * kk, lane), g[t]);
#pragma unroll
      for (int t = 4; t < 8; ++t)
        g[t] = wmma_bf16(a, frag_b_rows(Wfcb, 256, 16 * t, 32 * kk, lane), g[t]);
    }

    // ---- epilogue
    if (phase == 0) {
#pragma unroll
      for (int t = 0; t < 8; ++t) {
        float bias = b_fc[16 * t + col];
        float s1 = 0.f, s2 = 0.f;
#pragma unroll
        for (int r = 0; r < 8; ++r) {
          if (!((lane & 16) && r >= 4)) {    // rows 12..15 excluded from stats
            float x = g[t][r] + bias;
            s1 += x; s2 += x * x;
          }
        }
        s1 += __shfl_xor(s1, 16);
        s2 += __shfl_xor(s2, 16);
        if (lane < 16) {
          atomicAdd(&sStat[16 * t + col], s1);
          atomicAdd(&sStat[128 + 16 * t + col], s2);
        }
      }
    } else {
#pragma unroll
      for (int t = 0; t < 4; ++t) {
        int cf = 16 * t + col, cc = cf + 64;
        float bfv = b_fc[cf], bcv = b_fc[cc];
        float scf = bn1_scale[cf], shf = bn1_shift[cf];
        float scc = bn1_scale[cc], shc = bn1_shift[cc];
        float s1 = 0.f;
#pragma unroll
        for (int r = 0; r < 8; ++r) {
          float xf = (g[t][r]     + bfv) * scf + shf;   // filter (BN1 applied)
          float xc = (g[t + 4][r] + bcv) * scc + shc;   // core
          float sig = 1.f / (1.f + __expf(-xf));
          float sp  = (xc > 20.f) ? xc : log1pf(__expf(xc));
          if (!((lane & 16) && r >= 4)) s1 += sig * sp;
        }
        s1 += __shfl_xor(s1, 16);
        if (lane < 16) {
          nbr_out[n * 64 + cf] = s1;
          atomicAdd(&sStat[cf], s1);
          atomicAdd(&sStat[64 + cf], s1 * s1);
        }
      }
    }
  }

  __syncthreads();
  int nstat = (phase == 0) ? 256 : 128;
  for (int i = tid; i < nstat; i += 64) atomicAdd(&stats_out[i], sStat[i]);
}

// ---------------------------------------------------------------------------
// Weight f32->bf16 conversion + stat-accumulator zeroing (runs every call so
// graph replays are deterministic).
// ---------------------------------------------------------------------------
__global__ void prep_weights(const float* __restrict__ WK, const float* __restrict__ WQ,
                             const float* __restrict__ WV, const float* __restrict__ WO,
                             const float* __restrict__ Wfc,
                             bf16_t* __restrict__ WKb, bf16_t* __restrict__ WQb,
                             bf16_t* __restrict__ WVb, bf16_t* __restrict__ WOb,
                             bf16_t* __restrict__ Wfcb,
                             float* __restrict__ stats1, float* __restrict__ stats2)
{
  int gid = blockIdx.x * blockDim.x + threadIdx.x;
  int stride = gridDim.x * blockDim.x;
  for (int i = gid; i < 64 * 192; i += stride) {
    WKb[i] = (bf16_t)WK[i];
    WQb[i] = (bf16_t)WQ[i];
    WVb[i] = (bf16_t)WV[i];
  }
  for (int i = gid; i < 256 * 64; i += stride) WOb[i] = (bf16_t)WO[i];
  for (int i = gid; i < 128 * 256; i += stride) Wfcb[i] = (bf16_t)Wfc[i];
  if (gid < 256) stats1[gid] = 0.f;
  if (gid < 128) stats2[gid] = 0.f;
}

// stats = [sum[C], sumsq[C]] -> scale_shift = [scale[C], shift[C]]
__global__ void bn_finalize(const float* __restrict__ stats,
                            const float* __restrict__ gamma,
                            const float* __restrict__ beta,
                            float* __restrict__ scale_shift, int C, float inv_count)
{
  int c = blockIdx.x * blockDim.x + threadIdx.x;
  if (c < C) {
    float mean = stats[c] * inv_count;
    float var  = stats[C + c] * inv_count - mean * mean;
    float sc   = gamma[c] * rsqrtf(var + 1e-5f);
    scale_shift[c]     = sc;
    scale_shift[C + c] = beta[c] - mean * sc;
  }
}

// out = softplus(atom_in + BN2(nbr_sumed))
__global__ void final_output(const float* __restrict__ atom_in,
                             const float* __restrict__ nbr,
                             const float* __restrict__ bn2,
                             float* __restrict__ out, int total)
{
  int i = blockIdx.x * blockDim.x + threadIdx.x;
  if (i < total) {
    int c = i & 63;
    float x = atom_in[i] + nbr[i] * bn2[c] + bn2[64 + c];
    out[i] = (x > 20.f) ? x : log1pf(__expf(x));
  }
}

// ---------------------------------------------------------------------------
extern "C" void kernel_launch(void* const* d_in, const int* in_sizes, int n_in,
                              void* d_out, int out_size, void* d_ws, size_t ws_size,
                              hipStream_t stream) {
  (void)in_sizes; (void)n_in; (void)out_size; (void)ws_size;
  const float* atom_in = (const float*)d_in[0];
  const float* nbr_fea = (const float*)d_in[1];
  const int*   nbr_idx = (const int*)d_in[2];
  const float* WK   = (const float*)d_in[3];
  const float* WQ   = (const float*)d_in[4];
  const float* WV   = (const float*)d_in[5];
  const float* WO   = (const float*)d_in[6];
  const float* Wfc  = (const float*)d_in[7];
  const float* bfc  = (const float*)d_in[8];
  const float* bn1g = (const float*)d_in[9];
  const float* bn1b = (const float*)d_in[10];
  const float* bn2g = (const float*)d_in[11];
  const float* bn2b = (const float*)d_in[12];

  char* ws = (char*)d_ws;
  bf16_t* WKb   = (bf16_t*)(ws + 0);       // 64*192  bf16 = 24576 B
  bf16_t* WQb   = (bf16_t*)(ws + 24576);
  bf16_t* WVb   = (bf16_t*)(ws + 49152);
  bf16_t* WOb   = (bf16_t*)(ws + 73728);   // 256*64  bf16 = 32768 B
  bf16_t* Wfcb  = (bf16_t*)(ws + 106496);  // 128*256 bf16 = 65536 B
  float* stats1 = (float*)(ws + 172032);   // 256 f32
  float* bn1    = (float*)(ws + 173056);   // 256 f32 (scale|shift)
  float* stats2 = (float*)(ws + 174080);   // 128 f32
  float* bn2    = (float*)(ws + 174592);   // 128 f32 (scale|shift)
  float* nbrs   = (float*)(ws + 175104);   // 65536*64 f32 = 16 MB

  const int N = 65536;

  prep_weights<<<64, 256, 0, stream>>>(WK, WQ, WV, WO, Wfc,
                                       WKb, WQb, WVb, WOb, Wfcb, stats1, stats2);

  // phase 0: accumulate BN1 stats (bn1 scale/shift pointers unused this pass)
  conv_attn_fused<<<4096, 64, 0, stream>>>(atom_in, nbr_fea, nbr_idx,
                                           WKb, WQb, WVb, WOb, Wfcb, bfc,
                                           bn1, bn1 + 128, stats1, nbrs, N, 0);

  bn_finalize<<<1, 128, 0, stream>>>(stats1, bn1g, bn1b, bn1, 128,
                                     1.f / (float)(N * 12));

  // phase 1: recompute, apply BN1, gate, slot-sum, accumulate BN2 stats
  conv_attn_fused<<<4096, 64, 0, stream>>>(atom_in, nbr_fea, nbr_idx,
                                           WKb, WQb, WVb, WOb, Wfcb, bfc,
                                           bn1, bn1 + 128, stats2, nbrs, N, 1);

  bn_finalize<<<1, 64, 0, stream>>>(stats2, bn2g, bn2b, bn2, 64, 1.f / (float)N);

  final_output<<<(N * 64) / 256, 256, 0, stream>>>(atom_in, nbrs, bn2,
                                                   (float*)d_out, N * 64);
}
